// ColumnGraphMemory_33088428049181
// MI455X (gfx1250) — matrix operand order repaired and checked by reading, not compiled, not run on hardware
//
#include <hip/hip_runtime.h>
#include <hip/hip_bf16.h>

// ---------------- problem constants ----------------
static constexpr int BB   = 4;
static constexpr int NN   = 4096;
static constexpr int DSZ  = 256;
static constexpr int DIDZ = 64;
static constexpr int EE   = 32768;
static constexpr int VV   = 32000;
static constexpr int NIZ  = 256;
static constexpr int NOZ  = 256;
static constexpr int H_UP = 1024;
static constexpr int H_CT = 512;
static constexpr int H_DL = 512;
static constexpr int H_SC = 128;

// ---------------- workspace layout (bytes) ----------------
static constexpr size_t SZ_STATE  = (size_t)BB * NN * DSZ * sizeof(float);   // 16 MB
static constexpr size_t OFF_SWORK = 0;
static constexpr size_t OFF_MOUT  = OFF_SWORK + SZ_STATE;
static constexpr size_t OFF_INC   = OFF_MOUT  + SZ_STATE;
static constexpr size_t OFF_KIN   = OFF_INC   + SZ_STATE;
static constexpr size_t OFF_VIN   = OFF_KIN   + (size_t)BB * DSZ * sizeof(float);
static constexpr size_t OFF_MOTOR = OFF_VIN   + (size_t)BB * DSZ * sizeof(float);
static constexpr size_t OFF_CW1   = OFF_MOTOR + (size_t)BB * DSZ * sizeof(float);
static constexpr size_t OFF_CW2   = OFF_CW1 + (size_t)320  * H_CT * 2;
static constexpr size_t OFF_DW1   = OFF_CW2 + (size_t)H_CT * DSZ  * 2;
static constexpr size_t OFF_DW2   = OFF_DW1 + (size_t)320  * H_DL * 2;
static constexpr size_t OFF_SW1   = OFF_DW2 + (size_t)H_DL * DSZ  * 2;
static constexpr size_t OFF_UW1   = OFF_SW1 + (size_t)384  * H_SC * 2;
static constexpr size_t OFF_UW2   = OFF_UW1 + (size_t)576  * H_UP * 2;

// ---------------- vector types ----------------
typedef __attribute__((ext_vector_type(16))) __bf16         bf16x16;
typedef __attribute__((ext_vector_type(16))) unsigned short u16x16;
typedef __attribute__((ext_vector_type(8)))  unsigned short u16x8;
typedef __attribute__((ext_vector_type(4)))  unsigned short u16x4;
typedef __attribute__((ext_vector_type(8)))  float          v8f;
typedef __attribute__((ext_vector_type(4)))  int            v4i_t;

// ---------------- helpers ----------------
__device__ __forceinline__ unsigned short bf16bits(float f) {
    unsigned u = __builtin_bit_cast(unsigned, f);
    unsigned r = u + 0x7FFFu + ((u >> 16) & 1u);      // round-to-nearest-even
    return (unsigned short)(r >> 16);
}
__device__ __forceinline__ void store4bf(unsigned short* p, float4 v) {
    u16x4 o;
    o[0] = bf16bits(v.x); o[1] = bf16bits(v.y);
    o[2] = bf16bits(v.z); o[3] = bf16bits(v.w);
    *(u16x4*)p = o;
}
// branchless tanh-approx GELU (matches jax.nn.gelu approximate=True)
__device__ __forceinline__ float gelu_f(float x) {
    float u  = 0.7978845608028654f * (x + 0.044715f * x * x * x);
    float e  = __expf(2.f * u);
    float th = 1.f - 2.f / (e + 1.f);     // tanh(u), saturates at +-1
    return 0.5f * x * (1.f + th);
}
__device__ __forceinline__ float sigmoid_f(float x) { return 1.f / (1.f + __expf(-x)); }

// ---------------- CDNA5 async global->LDS (guarded) ----------------
#if __has_builtin(__builtin_amdgcn_global_load_async_to_lds_b128)
#define HAS_ASYNC_LDS 1
typedef __attribute__((address_space(1))) v4i_t* gas_v4i_p;   // global v4i*
typedef __attribute__((address_space(3))) v4i_t* las_v4i_p;   // LDS v4i*
__device__ __forceinline__ void async_copy16(const float* g, float* l) {
    // AS1 flat address == global address; low 32 bits of a flat LDS address == LDS offset
    gas_v4i_p gp = (gas_v4i_p)(unsigned long long)(uintptr_t)g;
    las_v4i_p lp = (las_v4i_p)(unsigned)(uintptr_t)l;
    __builtin_amdgcn_global_load_async_to_lds_b128(gp, lp, 0, 0);
}
#else
#define HAS_ASYNC_LDS 0
#endif
__device__ __forceinline__ void wait_async0() {
#if __has_builtin(__builtin_amdgcn_s_wait_asynccnt)
    __builtin_amdgcn_s_wait_asynccnt(0);
#else
    asm volatile("s_wait_asynccnt 0" ::: "memory");
#endif
}

// A-fragment: 16x32 bf16 tile from LDS (row-major, ld elements).
// lanes 0-15 row M=lane, K {0..7,16..23}; lanes 16-31 row M=lane-16, K {8..15,24..31}.
__device__ __forceinline__ bf16x16 frag_A_lds(const unsigned short* x, int ld, int k0) {
    int lane = threadIdx.x & 31;
    int row  = lane & 15;
    int kb   = k0 + ((lane & 16) ? 8 : 0);
    const unsigned short* p = x + row * ld + kb;
    u16x8 lo = *(const u16x8*)p;          // K kb..kb+7
    u16x8 hi = *(const u16x8*)(p + 16);   // K kb+16..kb+23
    u16x16 r;
#pragma unroll
    for (int j = 0; j < 8; ++j) { r[j] = lo[j]; r[j + 8] = hi[j]; }
    return __builtin_bit_cast(bf16x16, r);
}

// B-fragment from pre-swizzled bf16 weights: each lane reads 16 contiguous shorts.
__device__ __forceinline__ bf16x16 frag_B_sw(const unsigned short* Wsw, int Hdim, int k0, int n0) {
    int lane = threadIdx.x & 31;
    const unsigned short* p =
        Wsw + ((size_t)((k0 >> 5) * (Hdim >> 4) + (n0 >> 4)) << 9) + lane * 16;
    return __builtin_bit_cast(bf16x16, *(const u16x16*)p);
}

__device__ __forceinline__ v8f wmma_bf16(bf16x16 a, bf16x16 b, v8f c) {
    return __builtin_amdgcn_wmma_f32_16x16x32_bf16(false, a, false, b, (short)0, c, false, false);
}

// ---------------- utility kernels ----------------
__global__ void __launch_bounds__(256) k_zero(float4* __restrict__ p, int n4) {
    int i = blockIdx.x * 256 + threadIdx.x;
    if (i < n4) p[i] = make_float4(0.f, 0.f, 0.f, 0.f);
}

// Convert fp32 weight matrix (K x H, row-major) into frag-swizzled bf16.
__global__ void __launch_bounds__(256) k_cvt_sw(const float* __restrict__ W,
                                                unsigned short* __restrict__ out,
                                                int K, int H) {
    int idx = blockIdx.x * 256 + threadIdx.x;
    if (idx >= K * H) return;
    int j    = idx & 15;
    int lane = (idx >> 4) & 31;
    int tl   = idx >> 9;
    int ntil = H >> 4;
    int nt   = tl % ntil;
    int kt   = tl / ntil;
    int k    = (kt << 5) + ((lane & 16) ? 8 : 0) + (j & 7) + ((j & 8) ? 16 : 0);
    int col  = (nt << 4) + (lane & 15);
    out[idx] = bf16bits(W[(size_t)k * H + col]);
}

// ---------------- stage 1: k = h@Wk, v = h@Wv ----------------
__global__ void __launch_bounds__(256) k_kv(const int* __restrict__ token_id,
                                            const float* __restrict__ emb,
                                            const float* __restrict__ in_k_w,
                                            const float* __restrict__ in_v_w,
                                            float* __restrict__ kin,
                                            float* __restrict__ vin) {
    __shared__ float h[DSZ];
    int t = threadIdx.x, b = blockIdx.x;
    h[t] = emb[(size_t)token_id[b] * DSZ + t];
    __syncthreads();
    float kk = 0.f, vv = 0.f;
    for (int d = 0; d < DSZ; ++d) {
        float hv = h[d];
        kk += hv * in_k_w[d * DSZ + t];
        vv += hv * in_v_w[d * DSZ + t];
    }
    kin[b * DSZ + t] = kk;
    vin[b * DSZ + t] = vv;
}

// ---------------- stage 2: gated input write ----------------
__global__ void __launch_bounds__(256) k_input_gate(float* __restrict__ s_work,
                                                    const int* __restrict__ in_pos,
                                                    const float* __restrict__ col_id,
                                                    const float* __restrict__ in_norm_w,
                                                    const float* __restrict__ in_q_w,
                                                    const float* __restrict__ kin,
                                                    const float* __restrict__ vin) {
    __shared__ float red[256];
    __shared__ float xc[320];
    int t = threadIdx.x;
    int b = blockIdx.x >> 8;
    int i = blockIdx.x & 255;
    int n = in_pos[i];
    size_t base = ((size_t)b * NN + n) * DSZ;
    float sv = s_work[base + t];
    red[t] = sv * sv;
    __syncthreads();
    for (int s = 128; s > 0; s >>= 1) { if (t < s) red[t] += red[t + s]; __syncthreads(); }
    float inv = rsqrtf(red[0] * (1.f / DSZ) + 1e-6f);
    xc[t] = sv * inv * in_norm_w[t];
    if (t < DIDZ) xc[256 + t] = col_id[n * DIDZ + t];
    __syncthreads();
    float q = 0.f;
    for (int d = 0; d < 320; ++d) q += xc[d] * in_q_w[d * DSZ + t];
    red[t] = q * kin[b * DSZ + t];
    __syncthreads();
    for (int s = 128; s > 0; s >>= 1) { if (t < s) red[t] += red[t + s]; __syncthreads(); }
    float gate = sigmoid_f(red[0] * 0.0625f);   // scale = 1/sqrt(256)
    s_work[base + t] = sv + gate * vin[b * DSZ + t];
}

// ---------------- stage 3: content MLP (WMMA) ----------------
__global__ void __launch_bounds__(256) k_content(const float* __restrict__ s_work,
                                                 const float* __restrict__ col_id,
                                                 const float* __restrict__ norm_w,
                                                 const unsigned short* __restrict__ W1,
                                                 const float* __restrict__ b1,
                                                 const unsigned short* __restrict__ W2,
                                                 const float* __restrict__ b2,
                                                 float* __restrict__ m_out) {
    __shared__ __align__(16) unsigned short xb[16 * 320];
    __shared__ __align__(16) unsigned short hb[16 * H_CT];
    __shared__ float red[256];
    __shared__ float inv[16];
    int t = threadIdx.x, wave = t >> 5;
    int r0 = blockIdx.x * 16;
    {
        int r = t >> 4, seg = t & 15;
        const float4* p = (const float4*)(s_work + (size_t)(r0 + r) * DSZ) + seg * 4;
        float ss = 0.f;
#pragma unroll
        for (int j = 0; j < 4; ++j) {
            float4 v = p[j];
            ss += v.x * v.x + v.y * v.y + v.z * v.z + v.w * v.w;
        }
        red[t] = ss;
    }
    __syncthreads();
    if (t < 16) {
        float acc = 0.f;
#pragma unroll
        for (int j = 0; j < 16; ++j) acc += red[t * 16 + j];
        inv[t] = rsqrtf(acc * (1.f / DSZ) + 1e-6f);
    }
    __syncthreads();
#pragma unroll
    for (int it = 0; it < 4; ++it) {            // 16 rows x 64 float4 chunks
        int idx = t + 256 * it;
        int r = idx >> 6, c = idx & 63;
        float4 v = *(const float4*)(s_work + (size_t)(r0 + r) * DSZ + c * 4);
        float4 w = *(const float4*)(norm_w + c * 4);
        float sc = inv[r];
        float4 o = make_float4(v.x * sc * w.x, v.y * sc * w.y, v.z * sc * w.z, v.w * sc * w.w);
        store4bf(&xb[r * 320 + c * 4], o);
    }
    {                                           // 16 rows x 16 float4 chunks of col_id
        int r = t >> 4, c = t & 15;
        int n = (r0 + r) & (NN - 1);
        float4 v = *(const float4*)(col_id + n * DIDZ + c * 4);
        store4bf(&xb[r * 320 + 256 + c * 4], v);
    }
    __syncthreads();
    int lane = t & 31, col = lane & 15, rb = (lane & 16) ? 8 : 0;
    for (int tile = wave; tile < H_CT / 16; tile += 8) {
        int n0 = tile * 16;
        v8f acc = {};
        for (int k0 = 0; k0 < 320; k0 += 32)
            acc = wmma_bf16(frag_A_lds(xb, 320, k0), frag_B_sw(W1, H_CT, k0, n0), acc);
        float bias = b1[n0 + col];
#pragma unroll
        for (int i = 0; i < 8; ++i)
            hb[(rb + i) * H_CT + n0 + col] = bf16bits(gelu_f(acc[i] + bias));
    }
    __syncthreads();
    for (int tile = wave; tile < 16; tile += 8) {
        int n0 = tile * 16;
        v8f acc = {};
        for (int k0 = 0; k0 < H_CT; k0 += 32)
            acc = wmma_bf16(frag_A_lds(hb, H_CT, k0), frag_B_sw(W2, DSZ, k0, n0), acc);
        float bias = b2[n0 + col];
#pragma unroll
        for (int i = 0; i < 8; ++i)
            m_out[(size_t)(r0 + rb + i) * DSZ + n0 + col] = acc[i] + bias;
    }
}

// ---------------- stage 4: fused edge kernel (delta MLP + score MLP + scatter) ----------------
__global__ void __launch_bounds__(256) k_edge(const float* __restrict__ m_out,
                                              const float* __restrict__ s_work,
                                              const float* __restrict__ col_id,
                                              const int* __restrict__ edge_src,
                                              const int* __restrict__ edge_dst,
                                              const unsigned short* __restrict__ DW1,
                                              const float* __restrict__ db1,
                                              const unsigned short* __restrict__ DW2,
                                              const float* __restrict__ db2,
                                              const unsigned short* __restrict__ SW1,
                                              const float* __restrict__ sb1,
                                              const float* __restrict__ sw2,
                                              const float* __restrict__ sb2,
                                              const float* __restrict__ E_bias,
                                              float* __restrict__ incoming) {
    __shared__ __align__(16) unsigned short xd[16 * 320];
    __shared__ __align__(16) unsigned short xs[16 * 384];
    __shared__ __align__(16) unsigned short hd[16 * H_DL];
    __shared__ __align__(16) float msg[16 * 256];
#if HAS_ASYNC_LDS
    __shared__ __align__(16) float stmp[16 * 256];
#endif
    __shared__ float hs[16 * H_SC];
    __shared__ int   srcs[16], dsts[16];
    __shared__ float wgt[16];
    int t = threadIdx.x, wave = t >> 5;
    int b  = blockIdx.x >> 11;          // / (EE/16)
    int e0 = (blockIdx.x & 2047) << 4;
    if (t < 16) { srcs[t] = edge_src[e0 + t]; dsts[t] = edge_dst[e0 + t]; }
    if (t >= 32 && t < 48 && e0 + 16 < EE) {   // prefetch next tile's gather rows
        int ns = edge_src[e0 + 16 + (t - 32)];
        __builtin_prefetch(m_out  + ((size_t)b * NN + ns) * DSZ, 0, 3);
        __builtin_prefetch(s_work + ((size_t)b * NN + ns) * DSZ, 0, 3);
    }
    __syncthreads();
#if HAS_ASYNC_LDS
    // gather 16 rows of m_out and s_work straight into LDS via ASYNC loads
#pragma unroll
    for (int it = 0; it < 4; ++it) {
        int idx = t + 256 * it;                 // 1024 16-byte chunks
        int i = idx >> 6, c = idx & 63;
        size_t go = ((size_t)b * NN + srcs[i]) * DSZ + c * 4;
        async_copy16(m_out + go,  &msg[i * 256 + c * 4]);
        async_copy16(s_work + go, &stmp[i * 256 + c * 4]);
    }
#endif
    {                                           // col_id staging (16 rows x 16 float4)
        int i = t >> 4, c = t & 15;
        float4 cs = *(const float4*)(col_id + srcs[i] * DIDZ + c * 4);
        float4 cd = *(const float4*)(col_id + dsts[i] * DIDZ + c * 4);
        store4bf(&xd[i * 320 + 256 + c * 4], cd);
        store4bf(&xs[i * 384 + 256 + c * 4], cs);
        store4bf(&xs[i * 384 + 320 + c * 4], cd);
    }
#if HAS_ASYNC_LDS
    wait_async0();
    __syncthreads();
#pragma unroll
    for (int it = 0; it < 4; ++it) {            // LDS fp32 -> bf16 conversion
        int idx = t + 256 * it;
        int i = idx >> 6, c = idx & 63;
        float4 mv = *(const float4*)&msg[i * 256 + c * 4];
        float4 sv = *(const float4*)&stmp[i * 256 + c * 4];
        store4bf(&xd[i * 320 + c * 4], mv);
        store4bf(&xs[i * 384 + c * 4], sv);
    }
#else
#pragma unroll
    for (int it = 0; it < 4; ++it) {
        int idx = t + 256 * it;
        int i = idx >> 6, c = idx & 63;
        size_t go = ((size_t)b * NN + srcs[i]) * DSZ + c * 4;
        float4 mv = *(const float4*)(m_out + go);
        float4 sv = *(const float4*)(s_work + go);
        *(float4*)&msg[i * 256 + c * 4] = mv;
        store4bf(&xd[i * 320 + c * 4], mv);
        store4bf(&xs[i * 384 + c * 4], sv);
    }
#endif
    __syncthreads();
    int lane = t & 31, col = lane & 15, rb = (lane & 16) ? 8 : 0;
    // delta layer1 (320 -> 512)
    for (int tile = wave; tile < H_DL / 16; tile += 8) {
        int n0 = tile * 16;
        v8f acc = {};
        for (int k0 = 0; k0 < 320; k0 += 32)
            acc = wmma_bf16(frag_A_lds(xd, 320, k0), frag_B_sw(DW1, H_DL, k0, n0), acc);
        float bias = db1[n0 + col];
#pragma unroll
        for (int i = 0; i < 8; ++i)
            hd[(rb + i) * H_DL + n0 + col] = bf16bits(gelu_f(acc[i] + bias));
    }
    // score layer1 (384 -> 128): exactly one tile per wave
    {
        int n0 = wave * 16;
        v8f acc = {};
        for (int k0 = 0; k0 < 384; k0 += 32)
            acc = wmma_bf16(frag_A_lds(xs, 384, k0), frag_B_sw(SW1, H_SC, k0, n0), acc);
        float bias = sb1[n0 + col];
#pragma unroll
        for (int i = 0; i < 8; ++i)
            hs[(rb + i) * H_SC + n0 + col] = gelu_f(acc[i] + bias);
    }
    __syncthreads();
    // delta layer2 (512 -> 256), residual add into msg
    for (int tile = wave; tile < 16; tile += 8) {
        int n0 = tile * 16;
        v8f acc = {};
        for (int k0 = 0; k0 < H_DL; k0 += 32)
            acc = wmma_bf16(frag_A_lds(hd, H_DL, k0), frag_B_sw(DW2, DSZ, k0, n0), acc);
        float bias = db2[n0 + col];
#pragma unroll
        for (int i = 0; i < 8; ++i)
            msg[(rb + i) * 256 + n0 + col] += acc[i] + bias;
    }
    __syncthreads();
    if (t < 16) {
        float sc = sb2[0] + E_bias[e0 + t];
        for (int j = 0; j < H_SC; ++j) sc += hs[t * H_SC + j] * sw2[j];
        wgt[t] = sigmoid_f(sc);
    }
    __syncthreads();
#pragma unroll 4
    for (int i = 0; i < 16; ++i) {
        float v = wgt[i] * msg[i * 256 + t];
        atomicAdd(incoming + ((size_t)b * NN + dsts[i]) * DSZ + t, v);
    }
}

// ---------------- stage 5: update MLP (WMMA) + decay -> s_new ----------------
__global__ void __launch_bounds__(256) k_update(const float* __restrict__ s_work,
                                                const float* __restrict__ incoming,
                                                const float* __restrict__ col_id,
                                                const float* __restrict__ norm_s_w,
                                                const float* __restrict__ norm_in_w,
                                                const unsigned short* __restrict__ W1,
                                                const float* __restrict__ b1,
                                                const unsigned short* __restrict__ W2,
                                                const float* __restrict__ b2,
                                                const float* __restrict__ decay_w,
                                                const float* __restrict__ decay_b,
                                                float* __restrict__ s_new) {
    __shared__ __align__(16) unsigned short xb[16 * 576];
    __shared__ __align__(16) unsigned short hb[16 * H_UP];
    __shared__ float red[256];
    __shared__ float invs[16], invi[16], dec[16];
    int t = threadIdx.x, wave = t >> 5;
    int r0 = blockIdx.x * 16;
    float ss = 0.f, si = 0.f;
    {
        int r = t >> 4, seg = t & 15;
        const float4* ps = (const float4*)(s_work   + (size_t)(r0 + r) * DSZ) + seg * 4;
        const float4* pi = (const float4*)(incoming + (size_t)(r0 + r) * DSZ) + seg * 4;
#pragma unroll
        for (int j = 0; j < 4; ++j) {
            float4 a = ps[j]; ss += a.x * a.x + a.y * a.y + a.z * a.z + a.w * a.w;
            float4 c = pi[j]; si += c.x * c.x + c.y * c.y + c.z * c.z + c.w * c.w;
        }
    }
    red[t] = ss;
    __syncthreads();
    if (t < 16) {
        float acc = 0.f;
#pragma unroll
        for (int j = 0; j < 16; ++j) acc += red[t * 16 + j];
        invs[t] = rsqrtf(acc * (1.f / DSZ) + 1e-6f);
    }
    __syncthreads();
    red[t] = si;
    __syncthreads();
    if (t < 16) {
        float acc = 0.f;
#pragma unroll
        for (int j = 0; j < 16; ++j) acc += red[t * 16 + j];
        invi[t] = rsqrtf(acc * (1.f / DSZ) + 1e-6f);
        int n = (r0 + t) & (NN - 1);
        float d = decay_b[0];
        for (int j = 0; j < DIDZ; ++j) d += col_id[n * DIDZ + j] * decay_w[j];
        dec[t] = sigmoid_f(d);
    }
    __syncthreads();
#pragma unroll
    for (int it = 0; it < 4; ++it) {            // rms(s_work) part
        int idx = t + 256 * it;
        int r = idx >> 6, c = idx & 63;
        float4 v = *(const float4*)(s_work + (size_t)(r0 + r) * DSZ + c * 4);
        float4 w = *(const float4*)(norm_s_w + c * 4);
        float sc = invs[r];
        store4bf(&xb[r * 576 + c * 4],
                 make_float4(v.x * sc * w.x, v.y * sc * w.y, v.z * sc * w.z, v.w * sc * w.w));
    }
#pragma unroll
    for (int it = 0; it < 4; ++it) {            // rms(incoming) part
        int idx = t + 256 * it;
        int r = idx >> 6, c = idx & 63;
        float4 v = *(const float4*)(incoming + (size_t)(r0 + r) * DSZ + c * 4);
        float4 w = *(const float4*)(norm_in_w + c * 4);
        float sc = invi[r];
        store4bf(&xb[r * 576 + 256 + c * 4],
                 make_float4(v.x * sc * w.x, v.y * sc * w.y, v.z * sc * w.z, v.w * sc * w.w));
    }
    {                                           // col_id part
        int r = t >> 4, c = t & 15;
        int n = (r0 + r) & (NN - 1);
        float4 v = *(const float4*)(col_id + n * DIDZ + c * 4);
        store4bf(&xb[r * 576 + 512 + c * 4], v);
    }
    __syncthreads();
    int lane = t & 31, col = lane & 15, rb = (lane & 16) ? 8 : 0;
    for (int tile = wave; tile < H_UP / 16; tile += 8) {
        int n0 = tile * 16;
        v8f acc = {};
        for (int k0 = 0; k0 < 576; k0 += 32)
            acc = wmma_bf16(frag_A_lds(xb, 576, k0), frag_B_sw(W1, H_UP, k0, n0), acc);
        float bias = b1[n0 + col];
#pragma unroll
        for (int i = 0; i < 8; ++i)
            hb[(rb + i) * H_UP + n0 + col] = bf16bits(gelu_f(acc[i] + bias));
    }
    __syncthreads();
    for (int tile = wave; tile < 16; tile += 8) {
        int n0 = tile * 16;
        v8f acc = {};
        for (int k0 = 0; k0 < H_UP; k0 += 32)
            acc = wmma_bf16(frag_A_lds(hb, H_UP, k0), frag_B_sw(W2, DSZ, k0, n0), acc);
        int c = n0 + col;
        float bias = b2[c];
#pragma unroll
        for (int i = 0; i < 8; ++i) {
            int gr = r0 + rb + i;
            s_new[(size_t)gr * DSZ + c] =
                dec[rb + i] * s_work[(size_t)gr * DSZ + c] + acc[i] + bias;
        }
    }
}

// ---------------- stage 6: output attention -> motor ----------------
__global__ void __launch_bounds__(256) k_motor(const float* __restrict__ s_new,
                                               const int* __restrict__ out_pos,
                                               const float* __restrict__ out_norm_w,
                                               const float* __restrict__ out_k_w,
                                               const float* __restrict__ out_v_w,
                                               const float* __restrict__ motor_query,
                                               float* __restrict__ motor) {
    __shared__ float wkq[256];
    __shared__ float sc[256];
    __shared__ float red[256];
    __shared__ float y[256];
    __shared__ int   op[256];
    int t = threadIdx.x, b = blockIdx.x;
    op[t] = out_pos[t];
    float acc = 0.f;
    for (int j = 0; j < DSZ; ++j) acc += out_k_w[t * DSZ + j] * motor_query[j];
    wkq[t] = acc;
    __syncthreads();
    const float* row = s_new + ((size_t)b * NN + op[t]) * DSZ;
    float ss = 0.f;
    for (int d = 0; d < DSZ; ++d) { float v = row[d]; ss += v * v; }
    float inv = rsqrtf(ss * (1.f / DSZ) + 1e-6f);
    float sco = 0.f;
    for (int d = 0; d < DSZ; ++d) sco += row[d] * inv * out_norm_w[d] * wkq[d];
    sc[t] = sco * 0.0625f;
    __syncthreads();
    red[t] = sc[t];
    __syncthreads();
    for (int s = 128; s > 0; s >>= 1) { if (t < s) red[t] = fmaxf(red[t], red[t + s]); __syncthreads(); }
    float mx = red[0];
    __syncthreads();
    float e = __expf(sc[t] - mx);
    red[t] = e;
    __syncthreads();
    for (int s = 128; s > 0; s >>= 1) { if (t < s) red[t] += red[t + s]; __syncthreads(); }
    float attn = e / red[0];
    __syncthreads();
    sc[t] = attn;
    __syncthreads();
    float yy = 0.f;
    for (int i = 0; i < NOZ; ++i) yy += sc[i] * s_new[((size_t)b * NN + op[i]) * DSZ + t];
    y[t] = yy;
    __syncthreads();
    float mo = 0.f;
    for (int k = 0; k < DSZ; ++k) mo += y[k] * out_v_w[k * DSZ + t];
    motor[b * DSZ + t] = mo;
}

// ---------------- stage 7: logits = motor @ emb^T ----------------
__global__ void __launch_bounds__(256) k_logits(const float* __restrict__ motor,
                                                const float* __restrict__ emb,
                                                float* __restrict__ logits) {
    __shared__ float m[256];
    int t = threadIdx.x;
    int b = blockIdx.y;
    int v = blockIdx.x * 256 + t;
    m[t] = motor[b * DSZ + t];
    __syncthreads();
    const float4* row = (const float4*)(emb + (size_t)v * DSZ);
    float acc = 0.f;
    for (int d = 0; d < DSZ / 4; ++d) {
        float4 r4 = row[d];
        acc += m[d * 4] * r4.x + m[d * 4 + 1] * r4.y + m[d * 4 + 2] * r4.z + m[d * 4 + 3] * r4.w;
    }
    logits[(size_t)b * VV + v] = acc;
}

// ---------------- host-side orchestration ----------------
extern "C" void kernel_launch(void* const* d_in, const int* in_sizes, int n_in,
                              void* d_out, int out_size, void* d_ws, size_t ws_size,
                              hipStream_t stream) {
    (void)in_sizes; (void)n_in; (void)out_size; (void)ws_size;
    const int*   token_id  = (const int*)  d_in[0];
    const float* s         = (const float*)d_in[1];
    const float* E_bias    = (const float*)d_in[2];
    const int*   edge_src  = (const int*)  d_in[3];
    const int*   edge_dst  = (const int*)  d_in[4];
    const int*   in_pos    = (const int*)  d_in[5];
    const int*   out_pos   = (const int*)  d_in[6];
    const float* col_id    = (const float*)d_in[7];
    const float* motor_q   = (const float*)d_in[8];
    const float* emb       = (const float*)d_in[9];
    const float* in_norm_w = (const float*)d_in[10];
    const float* in_q_w    = (const float*)d_in[11];
    const float* in_k_w    = (const float*)d_in[12];
    const float* in_v_w    = (const float*)d_in[13];
    const float* c_norm_w  = (const float*)d_in[14];
    const float* c_w1      = (const float*)d_in[15];
    const float* c_b1      = (const float*)d_in[16];
    const float* c_w2      = (const float*)d_in[17];
    const float* c_b2      = (const float*)d_in[18];
    const float* dl_w1     = (const float*)d_in[19];
    const float* dl_b1     = (const float*)d_in[20];
    const float* dl_w2     = (const float*)d_in[21];
    const float* dl_b2     = (const float*)d_in[22];
    const float* sc_w1     = (const float*)d_in[23];
    const float* sc_b1     = (const float*)d_in[24];
    const float* sc_w2     = (const float*)d_in[25];
    const float* sc_b2     = (const float*)d_in[26];
    const float* u_norm_s  = (const float*)d_in[27];
    const float* u_norm_in = (const float*)d_in[28];
    const float* u_w1      = (const float*)d_in[29];
    const float* u_b1      = (const float*)d_in[30];
    const float* u_w2      = (const float*)d_in[31];
    const float* u_b2      = (const float*)d_in[32];
    const float* decay_w   = (const float*)d_in[33];
    const float* decay_b   = (const float*)d_in[34];
    const float* o_norm_w  = (const float*)d_in[35];
    const float* out_k_w   = (const float*)d_in[36];
    const float* out_v_w   = (const float*)d_in[37];

    char* ws = (char*)d_ws;
    float* s_work   = (float*)(ws + OFF_SWORK);
    float* m_out    = (float*)(ws + OFF_MOUT);
    float* incoming = (float*)(ws + OFF_INC);
    float* kin      = (float*)(ws + OFF_KIN);
    float* vin      = (float*)(ws + OFF_VIN);
    float* motor    = (float*)(ws + OFF_MOTOR);
    unsigned short* CW1 = (unsigned short*)(ws + OFF_CW1);
    unsigned short* CW2 = (unsigned short*)(ws + OFF_CW2);
    unsigned short* DW1 = (unsigned short*)(ws + OFF_DW1);
    unsigned short* DW2 = (unsigned short*)(ws + OFF_DW2);
    unsigned short* SW1 = (unsigned short*)(ws + OFF_SW1);
    unsigned short* UW1 = (unsigned short*)(ws + OFF_UW1);
    unsigned short* UW2 = (unsigned short*)(ws + OFF_UW2);

    float* logits = (float*)d_out;
    float* s_new  = (float*)d_out + (size_t)BB * VV;

    (void)hipMemcpyAsync(s_work, s, SZ_STATE, hipMemcpyDeviceToDevice, stream);
    k_zero<<<(BB * NN * DSZ / 4 + 255) / 256, 256, 0, stream>>>((float4*)incoming,
                                                                BB * NN * DSZ / 4);

    k_cvt_sw<<<(320  * H_CT) / 256, 256, 0, stream>>>(c_w1,  CW1, 320,  H_CT);
    k_cvt_sw<<<(H_CT * DSZ ) / 256, 256, 0, stream>>>(c_w2,  CW2, H_CT, DSZ);
    k_cvt_sw<<<(320  * H_DL) / 256, 256, 0, stream>>>(dl_w1, DW1, 320,  H_DL);
    k_cvt_sw<<<(H_DL * DSZ ) / 256, 256, 0, stream>>>(dl_w2, DW2, H_DL, DSZ);
    k_cvt_sw<<<(384  * H_SC) / 256, 256, 0, stream>>>(sc_w1, SW1, 384,  H_SC);
    k_cvt_sw<<<(576  * H_UP) / 256, 256, 0, stream>>>(u_w1,  UW1, 576,  H_UP);
    k_cvt_sw<<<(H_UP * DSZ ) / 256, 256, 0, stream>>>(u_w2,  UW2, H_UP, DSZ);

    k_kv<<<BB, 256, 0, stream>>>(token_id, emb, in_k_w, in_v_w, kin, vin);
    k_input_gate<<<BB * NIZ, 256, 0, stream>>>(s_work, in_pos, col_id, in_norm_w,
                                               in_q_w, kin, vin);
    k_content<<<BB * NN / 16, 256, 0, stream>>>(s_work, col_id, c_norm_w,
                                                CW1, c_b1, CW2, c_b2, m_out);
    k_edge<<<BB * EE / 16, 256, 0, stream>>>(m_out, s_work, col_id, edge_src, edge_dst,
                                             DW1, dl_b1, DW2, dl_b2,
                                             SW1, sc_b1, sc_w2, sc_b2, E_bias, incoming);
    k_update<<<BB * NN / 16, 256, 0, stream>>>(s_work, incoming, col_id,
                                               u_norm_s, u_norm_in,
                                               UW1, u_b1, UW2, u_b2,
                                               decay_w, decay_b, s_new);
    k_motor<<<BB, 256, 0, stream>>>(s_new, out_pos, o_norm_w, out_k_w, out_v_w,
                                    motor_q, motor);
    k_logits<<<dim3(VV / 256, BB), 256, 0, stream>>>(motor, emb, logits);
}
